// AttentionContextLayer_57844619542913
// MI455X (gfx1250) — matrix operand order repaired and checked by reading, not compiled, not run on hardware
//
#include <hip/hip_runtime.h>
#include <hip/hip_bf16.h>
#include <math.h>
#include <stdint.h>

typedef __attribute__((ext_vector_type(16))) _Float16 v16h;
typedef __attribute__((ext_vector_type(8)))  float    v8f;

#define B_  8
#define TQ_ 512
#define DQ_ 256
#define TV_ 256
#define DV_ 256
#define U_  128
#define QT_ 16          // query rows per fused block
#define UC_ 16          // u-chunk staged per TDM tile
#define NC_ (U_ / UC_)  // 8 chunks
#define VROW_ 20        // padded LDS row stride (16 data + 4 pad) -> 16B aligned, no bank conflicts

__device__ __forceinline__ float fast_tanh(float x) {
#if __has_builtin(__builtin_amdgcn_tanhf)
  return __builtin_amdgcn_tanhf(x);   // gfx1250 v_tanh_f32
#else
  return tanhf(x);
#endif
}

// ------------------------- Tensor Data Mover helpers -----------------------------
#if __has_builtin(__builtin_amdgcn_tensor_load_to_lds)
#define USE_TDM_ 1
typedef unsigned int tdm_v4u __attribute__((ext_vector_type(4)));
typedef int          tdm_v4i __attribute__((ext_vector_type(4)));
typedef int          tdm_v8i __attribute__((ext_vector_type(8)));

// DMA a [rows x cols] f32 tile (row stride `stride` elems) from global to LDS,
// with 4-dword padding every 16 dwords -> LDS row stride VROW_ floats.
__device__ __forceinline__ void tdm_load_tile(const float* gsrc, unsigned lds_byte,
                                              int rows, int cols, int stride) {
  unsigned long long ga = (unsigned long long)(uintptr_t)gsrc;
  tdm_v4u g0 = { 1u,                                   // count=1 valid descriptor
                 lds_byte,                             // lds_addr       [63:32]
                 (unsigned)(ga & 0xFFFFFFFFu),         // global_addr lo [95:64]
                 ((unsigned)((ga >> 32) & 0x01FFFFFFu))// global_addr hi [120:96]
                 | (2u << 30) };                       // type=2 (image) [127:126]
  tdm_v8i g1;
  g1[0] = (int)((2u << 16)       // data_size = 4 bytes
              | (1u << 20)       // pad_enable
              | (3u << 22)       // pad_interval: 16 dwords
              | (3u << 25));     // pad_amount: 4 dwords
  g1[1] = (int)(((unsigned)cols & 0xFFFFu) << 16);   // tensor_dim0[15:0] @ [63:48]
  g1[2] = (int)(((unsigned)rows & 0xFFFFu) << 16);   // tensor_dim1[15:0] @ [95:80]
  g1[3] = (int)(((unsigned)cols & 0xFFFFu) << 16);   // tile_dim0 @ [127:112]
  g1[4] = (int)((unsigned)rows & 0xFFFFu);           // tile_dim1 @ [143:128]
  g1[5] = (int)stride;                               // tensor_dim0_stride lo32
  g1[6] = 0; g1[7] = 0;
  tdm_v4i z4 = {0, 0, 0, 0};
#if __clang_major__ >= 23
  tdm_v8i z8 = {0, 0, 0, 0, 0, 0, 0, 0};
  __builtin_amdgcn_tensor_load_to_lds(g0, g1, z4, z4, z8, 0);
#else
  __builtin_amdgcn_tensor_load_to_lds(g0, g1, z4, z4, 0);
#endif
}

__device__ __forceinline__ void wait_tdm() {
#if __has_builtin(__builtin_amdgcn_s_wait_tensorcnt)
  __builtin_amdgcn_s_wait_tensorcnt((short)0);
#else
  asm volatile("s_wait_tensorcnt 0x0" ::: "memory");
#endif
}
#endif

// ---------------- K0: fold Wc = Wp@Wq [DQ,U], bc = bp@Wq + bq [U] ----------------
__global__ void fold_weights(const float* __restrict__ Wp, const float* __restrict__ bp,
                             const float* __restrict__ Wq, const float* __restrict__ bq,
                             float* __restrict__ Wc, float* __restrict__ bc) {
  int idx = blockIdx.x * blockDim.x + threadIdx.x;
  if (idx < DQ_ * U_) {
    int d = idx / U_, n = idx % U_;
    float s = 0.f;
#pragma unroll 8
    for (int u = 0; u < U_; ++u) s += Wp[d * U_ + u] * Wq[u * U_ + n];
    Wc[idx] = s;
  } else if (idx < DQ_ * U_ + U_) {
    int n = idx - DQ_ * U_;
    float s = bq[n];
    for (int u = 0; u < U_; ++u) s += bp[u] * Wq[u * U_ + n];
    bc[n] = s;
  }
}

// -------- K0b: pack W[K,N] f32 into fragment-linear f16 B-operand layout ---------
// P[((tn*(K/32)+kblk)*32 + lane)*16 + 2v + j] = (f16) W[kblk*32 + 2v + 8*(lane>>4) + 8*(v>>2) + j][tn*16 + (lane&15)]
__global__ void pack_w_f16(const float* __restrict__ W, _Float16* __restrict__ P,
                           int K, int N) {
  int idx = blockIdx.x * blockDim.x + threadIdx.x;
  if (idx >= K * N) return;
  const int kblocks = K >> 5;
  const int j    = idx & 1;
  const int v    = (idx >> 1) & 7;
  const int lane = (idx >> 4) & 31;
  const int kblk = (idx >> 9) % kblocks;
  const int tn   = (idx >> 9) / kblocks;
  const int k = (kblk << 5) + (v << 1) + ((lane >> 4) << 3) + ((v >> 2) << 3) + j;
  const int n = (tn << 4) + (lane & 15);
  P[idx] = (_Float16)W[(size_t)k * N + n];
}

// ------------- K1/K2: C[M,N] = A[M,K] @ W[K,N] + bias, f16 WMMA, f32 acc ---------
// One wave per 16x16 tile; B operand comes pre-packed (one 32B load per k-step).
__global__ void gemm_wmma(const float* __restrict__ A, const _Float16* __restrict__ Wp16,
                          const float* __restrict__ bias, float* __restrict__ C,
                          int M, int K, int N) {
  const int lane   = threadIdx.x & 31;
  const int wv     = threadIdx.x >> 5;
  const int tilesN = N >> 4;
  const int tile   = blockIdx.x * (blockDim.x >> 5) + wv;
  if (tile >= (M >> 4) * tilesN) return;
  const int tm = tile / tilesN, tn = tile % tilesN;
  const int row0 = tm << 4, col0 = tn << 4;
  const int mA   = lane & 15;
  const int half = lane >> 4;
  const int kblocks = K >> 5;

  const float*    arow = A + (size_t)(row0 + mA) * K + (half << 3);
  const _Float16* bpk  = Wp16 + (((size_t)tn * kblocks) << 9) + ((size_t)lane << 4);

  v8f c = {};
  for (int k0 = 0; k0 < K; k0 += 32) {
    const float4 a0 = *(const float4*)(arow + k0);
    const float4 a1 = *(const float4*)(arow + k0 + 4);
    const float4 a2 = *(const float4*)(arow + k0 + 16);
    const float4 a3 = *(const float4*)(arow + k0 + 20);
    v16h a;
    a[0]  = (_Float16)a0.x; a[1]  = (_Float16)a0.y; a[2]  = (_Float16)a0.z; a[3]  = (_Float16)a0.w;
    a[4]  = (_Float16)a1.x; a[5]  = (_Float16)a1.y; a[6]  = (_Float16)a1.z; a[7]  = (_Float16)a1.w;
    a[8]  = (_Float16)a2.x; a[9]  = (_Float16)a2.y; a[10] = (_Float16)a2.z; a[11] = (_Float16)a2.w;
    a[12] = (_Float16)a3.x; a[13] = (_Float16)a3.y; a[14] = (_Float16)a3.z; a[15] = (_Float16)a3.w;
    const v16h bf = *(const v16h*)(bpk + (((size_t)(k0 >> 5)) << 9));
    c = __builtin_amdgcn_wmma_f32_16x16x32_f16(false, a, false, bf, (short)0, c, false, false);
  }
  const float bval = bias[col0 + mA];
#pragma unroll
  for (int r = 0; r < 8; ++r) {
    const int m = r + (half << 3);
    C[(size_t)(row0 + m) * N + col0 + mA] = c[r] + bval;
  }
}

// ---------------- K3: fused energies + mask + softmax + context + concat ---------
__global__ void attn_fused(const float* __restrict__ qh, const float* __restrict__ vh,
                           const float* __restrict__ values, const int* __restrict__ mask,
                           const float* __restrict__ vvec, const float* __restrict__ vbp,
                           const float* __restrict__ query, float* __restrict__ out) {
  __shared__ float vhbuf[2][TV_ * VROW_];  // double-buffered TDM destination (20KB each)
  __shared__ float qbuf[2][TV_];           // qh chunk: [16 q][16 u], double-buffered
  __shared__ float vsc[U_];
  __shared__ float maskb[TV_];
  __shared__ float attn_lds[TV_];
  __shared__ float redbuf[8];

  const int t    = threadIdx.x;            // 0..255 : key index t / channel d
  const int b    = blockIdx.x >> 5;        // TQ_/QT_ = 32 tiles per batch
  const int qt   = blockIdx.x & 31;
  const int q0   = qt * QT_;
  const int lane = t & 31, wv = t >> 5;

  const float* vh_b = vh + (size_t)b * TV_ * U_;
  const size_t qrow_base = (size_t)(b * TQ_ + q0 + (t >> 4)) * U_ + (t & 15);

  if (t < U_) vsc[t] = vvec[t];
  maskb[t] = (1.0f - (float)mask[b * TV_ + t]) * -1e9f;
  const float vb0 = vbp[0];

  // ---- stage chunk 0 ----
#if defined(USE_TDM_)
  if (wv == 0)
    tdm_load_tile(vh_b, (unsigned)(uintptr_t)&vhbuf[0][0], TV_, UC_, U_);
#else
  for (int i = t; i < TV_ * UC_; i += 256) {
    const int tt = i >> 4, ul = i & 15;
    vhbuf[0][tt * VROW_ + ul] = vh_b[(size_t)tt * U_ + ul];
  }
#endif
  qbuf[0][t] = qh[qrow_base];
#if defined(USE_TDM_)
  if (wv == 0) wait_tdm();
#endif
  __syncthreads();

  float acc[QT_];
#pragma unroll
  for (int i = 0; i < QT_; ++i) acc[i] = 0.f;

  for (int c = 0; c < NC_; ++c) {
    const int cb = c & 1, nb = cb ^ 1;
    if (c + 1 < NC_) {   // prefetch next chunk while computing this one
#if defined(USE_TDM_)
      if (wv == 0)
        tdm_load_tile(vh_b + (c + 1) * UC_, (unsigned)(uintptr_t)&vhbuf[nb][0],
                      TV_, UC_, U_);
#else
      for (int i = t; i < TV_ * UC_; i += 256) {
        const int tt = i >> 4, ul = i & 15;
        vhbuf[nb][tt * VROW_ + ul] = vh_b[(size_t)tt * U_ + (c + 1) * UC_ + ul];
      }
#endif
      qbuf[nb][t] = qh[qrow_base + (c + 1) * UC_];
    }

    // thread-private vh row (reused across all 16 queries) + uniform v chunk
    const float* vr = &vhbuf[cb][t * VROW_];
    const float4 v0 = *(const float4*)(vr);
    const float4 v1 = *(const float4*)(vr + 4);
    const float4 v2 = *(const float4*)(vr + 8);
    const float4 v3 = *(const float4*)(vr + 12);
    const float4 s0 = *(const float4*)&vsc[c * UC_];
    const float4 s1 = *(const float4*)&vsc[c * UC_ + 4];
    const float4 s2 = *(const float4*)&vsc[c * UC_ + 8];
    const float4 s3 = *(const float4*)&vsc[c * UC_ + 12];

#pragma unroll
    for (int qq = 0; qq < QT_; ++qq) {
      const float* qp = &qbuf[cb][qq * UC_];
      const float4 q0v = *(const float4*)(qp);
      const float4 q1v = *(const float4*)(qp + 4);
      const float4 q2v = *(const float4*)(qp + 8);
      const float4 q3v = *(const float4*)(qp + 12);
      float a = acc[qq];
      a += s0.x * fast_tanh(q0v.x + v0.x); a += s0.y * fast_tanh(q0v.y + v0.y);
      a += s0.z * fast_tanh(q0v.z + v0.z); a += s0.w * fast_tanh(q0v.w + v0.w);
      a += s1.x * fast_tanh(q1v.x + v1.x); a += s1.y * fast_tanh(q1v.y + v1.y);
      a += s1.z * fast_tanh(q1v.z + v1.z); a += s1.w * fast_tanh(q1v.w + v1.w);
      a += s2.x * fast_tanh(q2v.x + v2.x); a += s2.y * fast_tanh(q2v.y + v2.y);
      a += s2.z * fast_tanh(q2v.z + v2.z); a += s2.w * fast_tanh(q2v.w + v2.w);
      a += s3.x * fast_tanh(q3v.x + v3.x); a += s3.y * fast_tanh(q3v.y + v3.y);
      a += s3.z * fast_tanh(q3v.z + v3.z); a += s3.w * fast_tanh(q3v.w + v3.w);
      acc[qq] = a;
    }
#if defined(USE_TDM_)
    if (wv == 0) wait_tdm();    // prefetched chunk done (overlapped with compute)
#endif
    __syncthreads();
  }

  // ---- mask + softmax + context, one query at a time ----
#pragma unroll
  for (int qq = 0; qq < QT_; ++qq) {
    float sc = acc[qq] + vb0 + maskb[t];
    float m = sc;
#pragma unroll
    for (int off = 16; off >= 1; off >>= 1) m = fmaxf(m, __shfl_xor(m, off, 32));
    if (lane == 0) redbuf[wv] = m;
    __syncthreads();
    float gm = redbuf[0];
#pragma unroll
    for (int i = 1; i < 8; ++i) gm = fmaxf(gm, redbuf[i]);
    const float e = __expf(sc - gm);
    float s = e;
#pragma unroll
    for (int off = 16; off >= 1; off >>= 1) s += __shfl_xor(s, off, 32);
    __syncthreads();
    if (lane == 0) redbuf[wv] = s;
    __syncthreads();
    float gs = 0.f;
#pragma unroll
    for (int i = 0; i < 8; ++i) gs += redbuf[i];
    attn_lds[t] = e / gs;
    __syncthreads();
    // context GEMV: thread t acts as output channel d = t (coalesced reads)
    float ca = 0.f;
#pragma unroll 4
    for (int tt = 0; tt < TV_; ++tt)
      ca += attn_lds[tt] * values[(size_t)(b * TV_ + tt) * DV_ + t];
    const int q = q0 + qq;
    out[(size_t)(b * TQ_ + q) * (DQ_ + DV_) + DQ_ + t] = ca;
    out[(size_t)(b * TQ_ + q) * (DQ_ + DV_) + t] =
        query[(size_t)(b * TQ_ + q) * DQ_ + t];
    __syncthreads();
  }
}

extern "C" void kernel_launch(void* const* d_in, const int* in_sizes, int n_in,
                              void* d_out, int out_size, void* d_ws, size_t ws_size,
                              hipStream_t stream) {
  (void)in_sizes; (void)n_in; (void)out_size; (void)ws_size;
  const float* query  = (const float*)d_in[0];
  const float* values = (const float*)d_in[1];
  const int*   mask   = (const int*)  d_in[2];
  const float* Wp     = (const float*)d_in[3];
  const float* bp     = (const float*)d_in[4];
  const float* Wq     = (const float*)d_in[5];
  const float* bq     = (const float*)d_in[6];
  const float* Wv     = (const float*)d_in[7];
  const float* bv     = (const float*)d_in[8];
  const float* vvec   = (const float*)d_in[9];
  const float* vb     = (const float*)d_in[10];
  float* out = (float*)d_out;

  // workspace: Wc | bc | qh | vh | Wc_packed(f16) | Wv_packed(f16)  (~3.4 MB)
  float* ws = (float*)d_ws;
  float* Wc = ws;
  float* bc = Wc + DQ_ * U_;
  float* qh = bc + U_;
  float* vh = qh + (size_t)B_ * TQ_ * U_;
  _Float16* Wcp = (_Float16*)(vh + (size_t)B_ * TV_ * U_);
  _Float16* Wvp = Wcp + DQ_ * U_;

  fold_weights<<<(DQ_ * U_ + U_ + 255) / 256, 256, 0, stream>>>(Wp, bp, Wq, bq, Wc, bc);
  pack_w_f16<<<(DQ_ * U_ + 255) / 256, 256, 0, stream>>>(Wc, Wcp, DQ_, U_);
  pack_w_f16<<<(DV_ * U_ + 255) / 256, 256, 0, stream>>>(Wv, Wvp, DV_, U_);

  {  // qh = query @ Wc + bc : [4096,256]x[256,128]
    const int M = B_ * TQ_, K = DQ_, N = U_;
    const int tiles = (M / 16) * (N / 16);
    gemm_wmma<<<tiles / 4, 128, 0, stream>>>(query, Wcp, bc, qh, M, K, N);
  }
  {  // vh = values @ Wv + bv : [2048,256]x[256,128]
    const int M = B_ * TV_, K = DV_, N = U_;
    const int tiles = (M / 16) * (N / 16);
    gemm_wmma<<<tiles / 4, 128, 0, stream>>>(values, Wvp, bv, vh, M, K, N);
  }

  attn_fused<<<B_ * (TQ_ / QT_), 256, 0, stream>>>(qh, vh, values, mask, vvec, vb,
                                                   query, out);
}
